// BertForEntityLinking1_85761906966873
// MI455X (gfx1250) — matrix operand order repaired and checked by reading, not compiled, not run on hardware
//
#include <hip/hip_runtime.h>
#include <hip/hip_bf16.h>
#include <math.h>

#define H_    768
#define S_    256
#define B_    32
#define G4H   3072      // 4*H
#define BS_   8192      // B*S
#define NWG_REC 32      // workgroups in recurrent kernel (each owns 24 hidden units)

typedef __attribute__((ext_vector_type(16))) __bf16      v16bf;
typedef __attribute__((ext_vector_type(8)))  float       v8f;
typedef __attribute__((ext_vector_type(4)))  unsigned    u32x4;
typedef __attribute__((ext_vector_type(4)))  unsigned    v4u;
typedef __attribute__((ext_vector_type(8)))  int         v8i;
typedef __attribute__((ext_vector_type(4)))  int         v4i;

union BF16Frag { v16bf v; u32x4 q[2]; };

__device__ __forceinline__ unsigned short f32_to_bf16(float f) {
  union { float f; unsigned u; } c; c.f = f;
  unsigned r = c.u + 0x7fffu + ((c.u >> 16) & 1u);
  return (unsigned short)(r >> 16);
}
__device__ __forceinline__ float sigmoidf_(float x) { return 1.0f / (1.0f + expf(-x)); }

// ---------------- grid-wide sense-reversing barrier (device-scope atomics) ---
__device__ __forceinline__ void grid_barrier(unsigned* cnt, unsigned* gen, unsigned nwg) {
  __syncthreads();
  if (threadIdx.x == 0) {
    __threadfence();  // release our stores to device scope
    unsigned g = __hip_atomic_load(gen, __ATOMIC_RELAXED, __HIP_MEMORY_SCOPE_AGENT);
    unsigned a = __hip_atomic_fetch_add(cnt, 1u, __ATOMIC_ACQ_REL, __HIP_MEMORY_SCOPE_AGENT);
    if (a == nwg - 1u) {
      __hip_atomic_store(cnt, 0u, __ATOMIC_RELAXED, __HIP_MEMORY_SCOPE_AGENT);
      __hip_atomic_fetch_add(gen, 1u, __ATOMIC_RELEASE, __HIP_MEMORY_SCOPE_AGENT);
    } else {
      while (__hip_atomic_load(gen, __ATOMIC_ACQUIRE, __HIP_MEMORY_SCOPE_AGENT) == g)
        __builtin_amdgcn_s_sleep(1);
    }
    __threadfence();  // acquire: invalidate so fresh h is observed
  }
  __syncthreads();
}

// ---------------- f32 -> bf16 conversion ------------------------------------
__global__ void __launch_bounds__(256)
cvt_bf16_kernel(const float* __restrict__ src, unsigned short* __restrict__ dst, int n) {
  int i = blockIdx.x * 256 + threadIdx.x;
  if (i < n) dst[i] = f32_to_bf16(src[i]);
}

// ---------------- embedding gather + bf16 convert ---------------------------
__global__ void __launch_bounds__(256)
embed_kernel(const int* __restrict__ ids, const float* __restrict__ tab,
             unsigned short* __restrict__ dst) {
  int i = blockIdx.x * 256 + threadIdx.x;     // grid sized exactly to BS_*H_
  int row = i / H_, h = i % H_;
  dst[i] = f32_to_bf16(tab[(long)ids[row] * H_ + h]);
}

// ---------------- input-gate GEMM:  G = X(8192x768) @ Wihᵀ(768x3072) --------
// 2Mx4N register blocking per wave: each B fragment feeds two WMMAs; A frags
// amortize over 4 N tiles. Block = 8 waves covering a 32x512 output panel.
__global__ void __launch_bounds__(256)
gemm_in_kernel(const unsigned short* __restrict__ X,
               const unsigned short* __restrict__ Wih,
               float* __restrict__ G) {
  const int lane = threadIdx.x & 31;
  const int wave = threadIdx.x >> 5;              // 8 waves
  const int m0   = blockIdx.x * 32;               // 256 M blocks
  const int n0   = blockIdx.y * 512 + wave * 64;  // 6 N groups * 8 waves * 64
  const int lm   = lane & 15;
  const bool hi  = lane >= 16;

  v8f acc[8] = {};
  const unsigned short* arow0 = X + (long)(m0 + lm) * H_ + (hi ? 8 : 0);
  const unsigned short* arow1 = arow0 + 16 * H_;

  for (int k0 = 0; k0 < H_; k0 += 32) {
    BF16Frag a0, a1;
    a0.q[0] = *(const u32x4*)(arow0 + k0);        // K 0..7   (or 8..15 hi)
    a0.q[1] = *(const u32x4*)(arow0 + k0 + 16);   // K 16..23 (or 24..31 hi)
    a1.q[0] = *(const u32x4*)(arow1 + k0);
    a1.q[1] = *(const u32x4*)(arow1 + k0 + 16);
#pragma unroll
    for (int t = 0; t < 4; ++t) {
      const unsigned short* brow =
          Wih + (long)(n0 + t * 16 + lm) * H_ + k0 + (hi ? 8 : 0);
      BF16Frag b;
      b.q[0] = *(const u32x4*)(brow);             // K 0..7   (8..15 hi)
      b.q[1] = *(const u32x4*)(brow + 16);        // K 16..23 (24..31 hi)
      acc[t]     = __builtin_amdgcn_wmma_f32_16x16x32_bf16(
          false, a0.v, false, b.v, (short)0, acc[t], false, false);
      acc[4 + t] = __builtin_amdgcn_wmma_f32_16x16x32_bf16(
          false, a1.v, false, b.v, (short)0, acc[4 + t], false, false);
    }
  }
#pragma unroll
  for (int t = 0; t < 4; ++t)
#pragma unroll
    for (int r = 0; r < 8; ++r) {
      int row0 = m0 + r + (hi ? 8 : 0);
      int col  = n0 + t * 16 + lm;
      G[(long)row0 * G4H + col]        = acc[t][r];
      G[(long)(row0 + 16) * G4H + col] = acc[4 + t][r];
    }
}

// ---------------- recurrent LSTM kernel (persistent, grid-synced) -----------
// NWG_REC=32 WGs; WG w owns hidden units j0=w*24 .. j0+23 -> gate columns
// {g*768 + j0 + jj}. w_hh slice (96x768 bf16 = 144KB) stays LDS-resident for
// all 256 timesteps (CDNA5 320KB LDS), staged via the Tensor Data Mover.
__global__ void __launch_bounds__(128)
lstm_rec_kernel(const float* __restrict__ gates,        // (BS_, 3072) f32
                const unsigned short* __restrict__ whh, // (3072,768) bf16
                const float* __restrict__ bih, const float* __restrict__ bhh,
                unsigned short* hbuf,                   // (32,768) bf16 shared state
                unsigned short* out_bf,                 // (BS_,768) bf16 or null
                float* out_f,                           // (BS_,768) f32 or null
                unsigned* bar_cnt, unsigned* bar_gen) {
  __shared__ unsigned short w_s[96 * H_];   // 147456 B
  __shared__ float Y_s[B_ * 96];            //  12288 B
  __shared__ float c_s[B_ * 24];            //   3072 B

  const int j0 = blockIdx.x * 24;

#if __has_builtin(__builtin_amdgcn_tensor_load_to_lds)
  // ---- TDM staging: 4 descriptors, one 24x768 bf16 tile per gate block ----
  if ((threadIdx.x >> 5) == 0) {            // one wave issues the DMAs
    unsigned ldsbase = (unsigned)(uintptr_t)(void*)w_s;   // low 32b = LDS offset
    unsigned long long gbase = (unsigned long long)(uintptr_t)whh;
#pragma unroll
    for (int g = 0; g < 4; ++g) {
      unsigned long long ga = gbase + (unsigned long long)(g * H_ + j0) * H_ * 2;
      v4u g0 = { 1u,                                        // count=1 (valid D#)
                 ldsbase + (unsigned)(g * 24 * H_ * 2),     // lds_addr [63:32]
                 (unsigned)(ga & 0xffffffffu),              // global_addr lo
                 (unsigned)((ga >> 32) & 0x01ffffffu) | (2u << 30) }; // hi + type=2
      v8i g1 = { (int)0x00010000,                 // data_size=1 (2 bytes)
                 (int)(H_ << 16),                 // tensor_dim0 lo16 @[63:48]
                 (int)((G4H & 0xffff) << 16),     // tensor_dim1 lo16 @[95:80]
                 (int)(H_ << 16),                 // tile_dim0 @[127:112]
                 24,                              // tile_dim1 @[143:128]
                 (int)H_,                         // tensor_dim0_stride lo32
                 0, 0 };
      v4i gz4 = { 0, 0, 0, 0 };
      v8i gz8 = { 0, 0, 0, 0, 0, 0, 0, 0 };
      __builtin_amdgcn_tensor_load_to_lds(g0, g1, gz4, gz4, gz8, 0);
    }
    __builtin_amdgcn_s_wait_tensorcnt(0);
  }
#else
  // ---- fallback: cooperative dword copy ----
  {
    const unsigned* src = (const unsigned*)whh;
    unsigned* dst = (unsigned*)w_s;
    for (int idx = threadIdx.x; idx < 96 * (H_ / 2); idx += 128) {
      int r = idx / (H_ / 2), d = idx % (H_ / 2);
      int Grow = (r / 24) * H_ + j0 + (r % 24);
      dst[r * (H_ / 2) + d] = src[(long)Grow * (H_ / 2) + d];
    }
  }
#endif
  // zero c state and our slice of h
  for (int e = threadIdx.x; e < B_ * 24; e += 128) {
    c_s[e] = 0.0f;
    int b = e / 24, j = e % 24;
    hbuf[b * H_ + j0 + j] = 0;
  }
  grid_barrier(bar_cnt, bar_gen, NWG_REC);  // also publishes LDS via syncthreads

  const int lane = threadIdx.x & 31;
  const int wave = threadIdx.x >> 5;        // 4 waves, 3 tiles each
  const int lm   = lane & 15;
  const bool hi  = lane >= 16;
  const int mt   = wave >> 1;               // M tile 0/1
  const int ntb  = (wave & 1) * 3;          // N tiles ntb..ntb+2 (of 6)
  const unsigned short* arow = hbuf + (mt * 16 + lm) * H_ + (hi ? 8 : 0);

  for (int t = 0; t < S_; ++t) {
    v8f acc[3] = {};
    for (int k0 = 0; k0 < H_; k0 += 32) {
      BF16Frag a;
      a.q[0] = *(const u32x4*)(arow + k0);
      a.q[1] = *(const u32x4*)(arow + k0 + 16);
#pragma unroll
      for (int i = 0; i < 3; ++i) {
        const unsigned short* brow =
            w_s + ((ntb + i) * 16 + lm) * H_ + k0 + (hi ? 8 : 0);
        BF16Frag b;
        b.q[0] = *(const u32x4*)(brow);
        b.q[1] = *(const u32x4*)(brow + 16);
        acc[i] = __builtin_amdgcn_wmma_f32_16x16x32_bf16(
            false, a.v, false, b.v, (short)0, acc[i], false, false);
      }
    }
#pragma unroll
    for (int i = 0; i < 3; ++i)
#pragma unroll
      for (int r = 0; r < 8; ++r) {
        int row = mt * 16 + r + (hi ? 8 : 0);
        Y_s[row * 96 + (ntb + i) * 16 + lm] = acc[i][r];
      }
    __syncthreads();

    // gate nonlinearities + state update for our 32x24 hidden slice
    for (int e = threadIdx.x; e < B_ * 24; e += 128) {
      int b = e / 24, j = e % 24;
      int gc = j0 + j;
      long grow = (long)(b * S_ + t) * G4H;
      float iv = sigmoidf_(Y_s[b * 96 + j]       + gates[grow + gc]            + bih[gc]            + bhh[gc]);
      float fv = sigmoidf_(Y_s[b * 96 + 24 + j]  + gates[grow + H_ + gc]       + bih[H_ + gc]       + bhh[H_ + gc]);
      float gv = tanhf   (Y_s[b * 96 + 48 + j]   + gates[grow + 2 * H_ + gc]   + bih[2 * H_ + gc]   + bhh[2 * H_ + gc]);
      float ov = sigmoidf_(Y_s[b * 96 + 72 + j]  + gates[grow + 3 * H_ + gc]   + bih[3 * H_ + gc]   + bhh[3 * H_ + gc]);
      float cc = fv * c_s[b * 24 + j] + iv * gv;
      c_s[b * 24 + j] = cc;
      float hh = ov * tanhf(cc);
      hbuf[b * H_ + gc] = f32_to_bf16(hh);
      long orow = (long)(b * S_ + t) * H_ + gc;
      if (out_bf) out_bf[orow] = f32_to_bf16(hh);
      if (out_f)  out_f[orow]  = hh;
    }
    grid_barrier(bar_cnt, bar_gen, NWG_REC);  // publish h for next step
  }
}

// ---------------- pool GEMM: seq_out(32x768) = Xflat(32x196608) @ pool_w ----
// Pure bandwidth (pool_w = 604 MB read once). f32 tiled FMA + prefetch-ahead.
__global__ void __launch_bounds__(256)
pool_kernel(const float* __restrict__ X, const float* __restrict__ W,
            const float* __restrict__ pb, float* __restrict__ seq_out) {
  __shared__ float Xs[B_][64];
  __shared__ float Ws[64][64];
  const int n0 = blockIdx.x * 64;     // 12 blocks
  const int b  = threadIdx.x >> 3;    // 0..31
  const int ng = threadIdx.x & 7;     // 0..7 -> 8 columns each
  float acc[8] = {};
  const int K = S_ * H_;              // 196608
  for (int k0 = 0; k0 < K; k0 += 64) {
    if (k0 + 64 < K && threadIdx.x < 64)   // prefetch next W chunk rows
      __builtin_prefetch(W + (long)(k0 + 64 + threadIdx.x) * H_ + n0, 0, 1);
    for (int e = threadIdx.x; e < B_ * 64; e += 256) {
      int bb = e >> 6, kk = e & 63;
      Xs[bb][kk] = X[(long)bb * K + k0 + kk];
    }
    for (int e = threadIdx.x; e < 64 * 64; e += 256) {
      int kk = e >> 6, nn = e & 63;
      Ws[kk][nn] = W[(long)(k0 + kk) * H_ + n0 + nn];
    }
    __syncthreads();
    for (int kk = 0; kk < 64; ++kk) {
      float xv = Xs[b][kk];
#pragma unroll
      for (int c = 0; c < 8; ++c) acc[c] += xv * Ws[kk][ng * 8 + c];
    }
    __syncthreads();
  }
#pragma unroll
  for (int c = 0; c < 8; ++c) {
    int n = n0 + ng * 8 + c;
    seq_out[b * H_ + n] = acc[c] + pb[n];
  }
}

// ---------------- entity max-pool + merge + final linear --------------------
__global__ void __launch_bounds__(256)
finalize_kernel(const float* __restrict__ out1, const int* __restrict__ entity,
                const float* __restrict__ seq_out, const float* __restrict__ lw,
                const float* __restrict__ lb, float* __restrict__ logits) {
  __shared__ float bo[H_];
  int b = blockIdx.x;
  for (int n = threadIdx.x; n < H_; n += 256) {
    float m = -INFINITY;
    for (int s = 0; s < S_; ++s) {
      float v = out1[(long)(b * S_ + s) * H_ + n];
      v = (entity[b * S_ + s] == 1) ? v : 0.0f;   // masked product incl. zeros
      m = fmaxf(m, v);
    }
    bo[n] = fmaxf(m, seq_out[b * H_ + n]);
  }
  __syncthreads();
  if (threadIdx.x < 2) {
    int t = threadIdx.x;
    float s = lb[t];
    for (int n = 0; n < H_; ++n) s += bo[n] * lw[n * 2 + t];
    logits[b * 2 + t] = s;
  }
}

// ---------------- BCEWithLogits mean ----------------------------------------
__global__ void __launch_bounds__(64)
loss_kernel(const float* __restrict__ logits, const int* __restrict__ y,
            float* __restrict__ loss_out) {
  __shared__ float red[64];
  int i = threadIdx.x;
  float l = logits[i];
  float yy = (float)y[i];
  red[i] = fmaxf(l, 0.0f) - l * yy + log1pf(expf(-fabsf(l)));
  __syncthreads();
  if (i == 0) {
    float s = 0.0f;
    for (int k = 0; k < 64; ++k) s += red[k];
    *loss_out = s / 64.0f;
  }
}

// ============================================================================
extern "C" void kernel_launch(void* const* d_in, const int* in_sizes, int n_in,
                              void* d_out, int out_size, void* d_ws, size_t ws_size,
                              hipStream_t stream) {
  const int*   input_ids = (const int*)  d_in[0];
  const int*   seq_label = (const int*)  d_in[3];
  const int*   entity    = (const int*)  d_in[4];
  const float* table     = (const float*)d_in[5];
  const float* w_ih0     = (const float*)d_in[6];
  const float* w_hh0     = (const float*)d_in[7];
  const float* b_ih0     = (const float*)d_in[8];
  const float* b_hh0     = (const float*)d_in[9];
  const float* w_ih1     = (const float*)d_in[10];
  const float* w_hh1     = (const float*)d_in[11];
  const float* b_ih1     = (const float*)d_in[12];
  const float* b_hh1     = (const float*)d_in[13];
  const float* pool_w    = (const float*)d_in[14];
  const float* pool_b    = (const float*)d_in[15];
  const float* lin_w     = (const float*)d_in[16];
  const float* lin_b     = (const float*)d_in[17];
  float* out = (float*)d_out;           // [loss, logits(64)]

  // ---- carve workspace (256B aligned) ----
  size_t off = 0;
  char* base = (char*)d_ws;
  auto carve = [&](size_t bytes) -> void* {
    void* p = base + off;
    off += (bytes + 255) & ~(size_t)255;
    return p;
  };
  const size_t WELEM = (size_t)G4H * H_;                 // 2359296
  unsigned short* wihb0 = (unsigned short*)carve(WELEM * 2);
  unsigned short* whhb0 = (unsigned short*)carve(WELEM * 2);
  unsigned short* wihb1 = (unsigned short*)carve(WELEM * 2);
  unsigned short* whhb1 = (unsigned short*)carve(WELEM * 2);
  unsigned short* xbf   = (unsigned short*)carve((size_t)BS_ * H_ * 2);
  unsigned short* out0b = (unsigned short*)carve((size_t)BS_ * H_ * 2);
  float* gates          = (float*)carve((size_t)BS_ * G4H * 4);   // reused both layers
  float* out1f          = (float*)carve((size_t)BS_ * H_ * 4);
  unsigned short* hbuf  = (unsigned short*)carve((size_t)B_ * H_ * 2);
  float* seqout         = (float*)carve((size_t)B_ * H_ * 4);
  unsigned* bar         = (unsigned*)carve(256);                  // [cnt, gen]

  (void)hipMemsetAsync(bar, 0, 256, stream);

  // bf16 weight copies
  const int wn = (int)WELEM, wb = (wn + 255) / 256;
  cvt_bf16_kernel<<<wb, 256, 0, stream>>>(w_ih0, wihb0, wn);
  cvt_bf16_kernel<<<wb, 256, 0, stream>>>(w_hh0, whhb0, wn);
  cvt_bf16_kernel<<<wb, 256, 0, stream>>>(w_ih1, wihb1, wn);
  cvt_bf16_kernel<<<wb, 256, 0, stream>>>(w_hh1, whhb1, wn);

  // embeddings -> bf16
  embed_kernel<<<(BS_ * H_) / 256, 256, 0, stream>>>(input_ids, table, xbf);

  dim3 ggrid(BS_ / 32, G4H / 512);      // (256, 6)

  // ---- layer 0 ----
  gemm_in_kernel<<<ggrid, 256, 0, stream>>>(xbf, wihb0, gates);
  lstm_rec_kernel<<<NWG_REC, 128, 0, stream>>>(gates, whhb0, b_ih0, b_hh0,
                                               hbuf, out0b, nullptr,
                                               bar, bar + 1);
  // ---- layer 1 ----
  gemm_in_kernel<<<ggrid, 256, 0, stream>>>(out0b, wihb1, gates);
  lstm_rec_kernel<<<NWG_REC, 128, 0, stream>>>(gates, whhb1, b_ih1, b_hh1,
                                               hbuf, nullptr, out1f,
                                               bar, bar + 1);

  // ---- pool + finalize + loss ----
  pool_kernel<<<H_ / 64, 256, 0, stream>>>(out1f, pool_w, pool_b, seqout);
  finalize_kernel<<<B_, 256, 0, stream>>>(out1f, entity, seqout, lin_w, lin_b, out + 1);
  loss_kernel<<<1, 64, 0, stream>>>(out + 1, seq_label, out);
  (void)in_sizes; (void)n_in; (void)out_size; (void)ws_size;
}
